// Model_GRU4REC_27333171872580
// MI455X (gfx1250) — matrix operand order repaired and checked by reading, not compiled, not run on hardware
//
#include <hip/hip_runtime.h>

// Problem constants (match reference)
#define B_TOT  1024
#define T_LEN  256
#define D_IN   64
#define H_DIM  128
#define K_DIM  192      // D + H
#define N_GATE 256      // 2H
#define TILE_B 16
#define LDA    200      // padded f16 row stride for the A tile (multiple of 8 -> 16B aligned chunks)

typedef _Float16 v8h  __attribute__((ext_vector_type(8)));
typedef _Float16 v16h __attribute__((ext_vector_type(16)));
typedef float    v8f  __attribute__((ext_vector_type(8)));

union F16Frag { v16h v; v8h h[2]; };

// A-matrix fragment per ISA 7.12.2 (16-bit A 16x32):
//   lanes 0-15 : row=lane,    elems 0..7 = K[k0..k0+7],   elems 8..15 = K[k0+16..k0+23]
//   lanes 16-31: row=lane-16, elems 0..7 = K[k0+8..+15],  elems 8..15 = K[k0+24..+31]
__device__ __forceinline__ v16h load_a_frag(const _Float16* sA, int row, int k0, int kh) {
    F16Frag f;
    f.h[0] = *(const v8h*)(sA + row * LDA + k0 + 8 * kh);
    f.h[1] = *(const v8h*)(sA + row * LDA + k0 + 16 + 8 * kh);
    return f.v;
}

// Single v_rcp_f32 instead of the IEEE divide expansion: the GRU elementwise path
// is on the serial critical path of the recurrence, keep it to a handful of VALU ops.
__device__ __forceinline__ float fast_rcp(float x)     { return __builtin_amdgcn_rcpf(x); }
__device__ __forceinline__ float fast_sigmoid(float x) { return fast_rcp(1.0f + __expf(-x)); }
__device__ __forceinline__ float fast_tanh(float x)    { return 1.0f - 2.0f * fast_rcp(1.0f + __expf(2.0f * x)); }

__global__ __launch_bounds__(256, 1)
void gru4rec_persistent_wmma(const int* __restrict__ item_his,   // [B, T]
                             const int* __restrict__ seq_lens,   // [B]
                             const float* __restrict__ emb,      // [N_MID, D]
                             const float* __restrict__ gk,       // [K, 2H]
                             const float* __restrict__ gb,       // [2H]
                             const float* __restrict__ ck,       // [K, H]
                             const float* __restrict__ cb,       // [H]
                             float* __restrict__ out) {          // [B, H]
    __shared__ __align__(16) _Float16 sA[TILE_B * LDA];   // [x_t | h] (f16), then h-part becomes r*h
    __shared__ float sH[TILE_B * H_DIM];                  // hidden state, fp32
    __shared__ float sU[TILE_B * H_DIM];                  // update gate u, fp32
    __shared__ int   sIdx[TILE_B * T_LEN];                // staged item indices for the whole tile
    __shared__ int   sLen[TILE_B];

    const int tid  = threadIdx.x;
    const int wave = tid >> 5;       // 0..7
    const int lane = tid & 31;
    const int lm   = lane & 15;
    const int kh   = lane >> 4;      // lane half selects K sub-range
    const int b0   = blockIdx.x * TILE_B;

    // ---- Preload weights into register-resident WMMA B fragments (transposed per-lane read).
    // B-matrix layout (matches A): lane = output column (lm); lanes 0-15 hold K[0..7],K[16..23],
    // lanes 16-31 hold K[8..15],K[24..31] of each 32-K step.
    v16h Bg0[6], Bg1[6], Bc[6];
#pragma unroll
    for (int kt = 0; kt < 6; ++kt) {
        const int kb = kt * 32 + kh * 8;
        v16h v0, v1, vc;
#pragma unroll
        for (int e = 0; e < 16; ++e) {
            v0[e] = (_Float16)gk[(size_t)(kb + (e < 8 ? e : e + 8)) * N_GATE + (wave * 32 + lm)];
            v1[e] = (_Float16)gk[(size_t)(kb + (e < 8 ? e : e + 8)) * N_GATE + (wave * 32 + 16 + lm)];
            vc[e] = (_Float16)ck[(size_t)(kb + (e < 8 ? e : e + 8)) * H_DIM  + (wave * 16 + lm)];
        }
        Bg0[kt] = v0; Bg1[kt] = v1; Bc[kt] = vc;
    }
    const float biasG0 = gb[wave * 32 + lm];
    const float biasG1 = gb[wave * 32 + 16 + lm];
    const float biasC  = cb[wave * 16 + lm];

    // ---- Init h = 0, stage sequence lengths and the whole index tile (coalesced).
    for (int i = tid; i < TILE_B * H_DIM; i += 256) sH[i] = 0.0f;
    for (int i = tid; i < TILE_B * T_LEN; i += 256) {
        const int m = i / T_LEN;
        const int t = i - m * T_LEN;
        sIdx[m * T_LEN + t] = item_his[(size_t)(b0 + m) * T_LEN + t];
    }
    if (tid < TILE_B) sLen[tid] = seq_lens[b0 + tid];
    __syncthreads();

    // ---- Recurrence over T steps.
    for (int t = 0; t < T_LEN; ++t) {
        // Stage A = [x_t | h] as f16 (embedding gather + h conversion).
        for (int i = tid; i < TILE_B * K_DIM; i += 256) {
            const int m = i / K_DIM;
            const int k = i - m * K_DIM;
            float v;
            if (k < D_IN) {
                const int idx = sIdx[m * T_LEN + t];
                v = emb[(size_t)idx * D_IN + k];
            } else {
                v = sH[m * H_DIM + (k - D_IN)];
            }
            sA[m * LDA + k] = (_Float16)v;
        }
        // Prefetch next step's embedding rows into cache (global_prefetch_b8);
        // overlaps the HBM gather latency with this step's GEMMs.
        if (t + 1 < T_LEN && tid < TILE_B) {
            const int nidx = sIdx[tid * T_LEN + t + 1];
            const float* p = emb + (size_t)nidx * D_IN;   // 256-byte row
            __builtin_prefetch(p, 0, 1);
            __builtin_prefetch(p + 32, 0, 1);             // second 128B line
        }
        __syncthreads();

        // Gate GEMM: [16 x 192] x [192 x 256]; wave w owns cols 32w..32w+31 (two N tiles).
        v8f accG0 = {};
        v8f accG1 = {};
#pragma unroll
        for (int kt = 0; kt < 6; ++kt) {
            const v16h a = load_a_frag(sA, lm, kt * 32, kh);
            accG0 = __builtin_amdgcn_wmma_f32_16x16x32_f16(false, a, false, Bg0[kt],
                                                           (short)0, accG0, false, false);
            accG1 = __builtin_amdgcn_wmma_f32_16x16x32_f16(false, a, false, Bg1[kt],
                                                           (short)0, accG1, false, false);
        }

        // C/D layout: elem e <-> (row m = e + 8*kh, col n = lm) within the 16x16 tile.
        if (wave < 4) {
            // Reset gate r (gate cols 0..127): overwrite h-columns of A with r*h.
            const int n0 = wave * 32 + lm;
            const int n1 = wave * 32 + 16 + lm;
#pragma unroll
            for (int e = 0; e < 8; ++e) {
                const int m = e + 8 * kh;
                const float r0 = fast_sigmoid(accG0[e] + biasG0);
                const float r1 = fast_sigmoid(accG1[e] + biasG1);
                sA[m * LDA + D_IN + n0] = (_Float16)(r0 * sH[m * H_DIM + n0]);
                sA[m * LDA + D_IN + n1] = (_Float16)(r1 * sH[m * H_DIM + n1]);
            }
        } else {
            // Update gate u (gate cols 128..255): stash in LDS for the h update.
            const int n0 = (wave - 4) * 32 + lm;
            const int n1 = (wave - 4) * 32 + 16 + lm;
#pragma unroll
            for (int e = 0; e < 8; ++e) {
                const int m = e + 8 * kh;
                sU[m * H_DIM + n0] = fast_sigmoid(accG0[e] + biasG0);
                sU[m * H_DIM + n1] = fast_sigmoid(accG1[e] + biasG1);
            }
        }
        __syncthreads();

        // Candidate GEMM: [16 x 192] x [192 x 128]; wave w owns cols 16w..16w+15.
        v8f accC = {};
#pragma unroll
        for (int kt = 0; kt < 6; ++kt) {
            const v16h a = load_a_frag(sA, lm, kt * 32, kh);
            accC = __builtin_amdgcn_wmma_f32_16x16x32_f16(false, a, false, Bc[kt],
                                                          (short)0, accC, false, false);
        }

        // h_new = u*h + (1-u)*tanh(c), with copy-through mask t < seq_len.
        {
            const int n = wave * 16 + lm;
#pragma unroll
            for (int e = 0; e < 8; ++e) {
                const int m  = e + 8 * kh;
                const float c  = fast_tanh(accC[e] + biasC);
                const float u  = sU[m * H_DIM + n];
                const float ho = sH[m * H_DIM + n];
                const float hn = u * ho + (1.0f - u) * c;
                if (t < sLen[m]) sH[m * H_DIM + n] = hn;
            }
        }
        __syncthreads();
    }

    // ---- Emit final hidden state.
    for (int i = tid; i < TILE_B * H_DIM; i += 256) {
        const int m = i / H_DIM;
        const int n = i - m * H_DIM;
        out[(size_t)(b0 + m) * H_DIM + n] = sH[i];
    }
}

extern "C" void kernel_launch(void* const* d_in, const int* in_sizes, int n_in,
                              void* d_out, int out_size, void* d_ws, size_t ws_size,
                              hipStream_t stream) {
    const int*   item_his = (const int*)d_in[0];
    const int*   seq_lens = (const int*)d_in[1];
    const float* emb      = (const float*)d_in[2];
    const float* gk       = (const float*)d_in[3];
    const float* gb       = (const float*)d_in[4];
    const float* ck       = (const float*)d_in[5];
    const float* cb       = (const float*)d_in[6];
    float*       out      = (float*)d_out;

    gru4rec_persistent_wmma<<<B_TOT / TILE_B, 256, 0, stream>>>(
        item_his, seq_lens, emb, gk, gb, ck, cb, out);
}
